// _FusedLoraWeightedLinear_8890582303287
// MI455X (gfx1250) — matrix operand order repaired and checked by reading, not compiled, not run on hardware
//
#include <hip/hip_runtime.h>
#include <hip/hip_bf16.h>

typedef unsigned short u16;
typedef __attribute__((ext_vector_type(16))) __bf16 bf16x16;
typedef __attribute__((ext_vector_type(8)))  __bf16 bf16x8;
typedef __attribute__((ext_vector_type(4)))  __bf16 bf16x4;
typedef __attribute__((ext_vector_type(8)))  float  f32x8;

#define DIM_M 8192   // B*S
#define DIM_N 4096   // O
#define DIM_K 4096   // D
#define RANKK 256    // E*R

#define BM 128
#define BN 128
#define BK 32
#define NT (DIM_K / BK)
#define LDP 40       // padded LDS row stride (elements): 80B rows -> conflict-free, 16B aligned

// Build a 16-element bf16 WMMA fragment from two contiguous 8-element LDS chunks.
__device__ __forceinline__ bf16x16 ld_frag(const u16* p, int hi_off) {
  bf16x8 lo = *(const bf16x8*)(p);
  bf16x8 hi = *(const bf16x8*)(p + hi_off);
  return __builtin_shufflevector(lo, hi, 0,1,2,3,4,5,6,7,8,9,10,11,12,13,14,15);
}

// Generic pointer to LDS: low 32 bits are the LDS byte address (ISA aperture rule).
__device__ __forceinline__ unsigned lds_addr(const void* p) {
  return (unsigned)(size_t)p;
}

// CDNA5 async memory->LDS copy, 16 bytes per lane. Tracked by ASYNCcnt.
__device__ __forceinline__ void async_load_b128(unsigned lds_off, const void* gaddr) {
  asm volatile("global_load_async_to_lds_b128 %0, %1, off"
               :: "v"(lds_off), "v"(gaddr)
               : "memory");
}

__device__ __forceinline__ void wait_async0() {
#if __has_builtin(__builtin_amdgcn_s_wait_asynccnt)
  __builtin_amdgcn_s_wait_asynccnt(0);
#else
  asm volatile("s_wait_asynccnt 0x0" ::: "memory");
#endif
}

// ---------------------------------------------------------------------------
// Kernel 1: build Bcat [O,256] (w_e * Bf[e,o,r]) and At [D,256] (A[e,r,d]),
// both bf16, K-contiguous so the GEMM fragments need no transpose.
// ---------------------------------------------------------------------------
__global__ void prep_factors(const float* __restrict__ A,
                             const float* __restrict__ Bfac,
                             const float* __restrict__ coeffs,
                             const float* __restrict__ scales,
                             u16* __restrict__ Bc, u16* __restrict__ At)
{
  const int half = 4096 * 32;                 // threads per half (8 k's each)
  int t = blockIdx.x * blockDim.x + threadIdx.x;
  if (t < half) {
    const int o  = t & 4095;
    const int kb = t >> 12;                   // 0..31
    bf16x8 chunk;
    #pragma unroll
    for (int j = 0; j < 8; ++j) {
      int k = kb * 8 + j;
      int e = k >> 4, r = k & 15;
      float w = coeffs[e] * scales[e];
      chunk[j] = (__bf16)(w * Bfac[(size_t)e * 4096 * 16 + o * 16 + r]);
    }
    *(bf16x8*)(&Bc[(size_t)o * RANKK + kb * 8]) = chunk;
  } else {
    int u = t - half;
    const int d  = u & 4095;
    const int kb = u >> 12;
    bf16x8 chunk;
    #pragma unroll
    for (int j = 0; j < 8; ++j) {
      int k = kb * 8 + j;
      int e = k >> 4, r = k & 15;
      chunk[j] = (__bf16)(A[((size_t)e * 16 + r) * 4096 + d]);
    }
    *(bf16x8*)(&At[(size_t)d * RANKK + kb * 8]) = chunk;
  }
}

// ---------------------------------------------------------------------------
// Kernel 2: W_eff[o,d] = bf16( W[o,d] + Bcat[o,:] . At[d,:] )  (K = 256)
// ---------------------------------------------------------------------------
__global__ __launch_bounds__(256)
void fold_wmma(const u16* __restrict__ Bc, const u16* __restrict__ At,
               const float* __restrict__ W, u16* __restrict__ Wb)
{
  __shared__ u16 Ls[BM * LDP];
  __shared__ u16 Rs[BN * LDP];
  const int t = threadIdx.x;
  const int lane = t & 31, wave = t >> 5;
  const int wm = wave >> 2, wn = wave & 3;
  const int lhalf = lane >> 4, l16 = lane & 15;
  const int bm0 = blockIdx.y * BM, bn0 = blockIdx.x * BN;

  f32x8 acc[4][2];
  for (int i = 0; i < 4; ++i)
    for (int j = 0; j < 2; ++j)
      for (int r = 0; r < 8; ++r) acc[i][j][r] = 0.f;

  for (int k0 = 0; k0 < RANKK; k0 += BK) {
    __syncthreads();
    #pragma unroll
    for (int p = 0; p < 2; ++p) {
      int row = (t >> 2) + p * 64;
      int c8  = (t & 3) << 3;
      *(uint4*)(&Ls[row * LDP + c8]) = *(const uint4*)(&Bc[(size_t)(bm0 + row) * RANKK + k0 + c8]);
      *(uint4*)(&Rs[row * LDP + c8]) = *(const uint4*)(&At[(size_t)(bn0 + row) * RANKK + k0 + c8]);
    }
    __syncthreads();

    bf16x16 afrag[4], bfrag[2];
    #pragma unroll
    for (int i = 0; i < 4; ++i)
      afrag[i] = ld_frag(&Ls[(wm * 64 + i * 16 + l16) * LDP + lhalf * 8], 16);
    #pragma unroll
    for (int j = 0; j < 2; ++j)
      bfrag[j] = ld_frag(&Rs[(wn * 32 + j * 16 + l16) * LDP + lhalf * 8], 16);

    #pragma unroll
    for (int i = 0; i < 4; ++i)
      #pragma unroll
      for (int j = 0; j < 2; ++j)
        acc[i][j] = __builtin_amdgcn_wmma_f32_16x16x32_bf16(
            false, afrag[i], false, bfrag[j], (short)0, acc[i][j], false, false);
  }

  #pragma unroll
  for (int i = 0; i < 4; ++i) {
    int m = bm0 + wm * 64 + i * 16 + lhalf * 8;
    #pragma unroll
    for (int j = 0; j < 2; ++j) {
      int n = bn0 + wn * 32 + j * 16 + l16;
      #pragma unroll
      for (int r = 0; r < 8; ++r) {
        size_t idx = (size_t)(m + r) * DIM_N + n;
        __bf16 hv = (__bf16)(W[idx] + acc[i][j][r]);
        Wb[idx] = __builtin_bit_cast(u16, hv);
      }
    }
  }
}

// ---------------------------------------------------------------------------
// Kernel 3: out[m,n] = x[m,:] . W_eff[n,:] + bias[n]
//   - double-buffered LDS pipeline
//   - W_eff tile: GLOBAL_LOAD_ASYNC_TO_LDS_B128 (ASYNCcnt), no VGPR round-trip
//   - x tile: global->VGPR fp32, cvt bf16, ds_store (issued around the WMMAs)
// ---------------------------------------------------------------------------
__global__ __launch_bounds__(256)
void main_wmma(const float* __restrict__ X, const u16* __restrict__ Wb,
               const float* __restrict__ bias, float* __restrict__ Out)
{
  __shared__ u16 Xs[2][BM * LDP];
  __shared__ u16 Ws[2][BN * LDP];
  const int t = threadIdx.x;
  const int lane = t & 31, wave = t >> 5;
  const int wm = wave >> 2, wn = wave & 3;
  const int lhalf = lane >> 4, l16 = lane & 15;
  const int bm0 = blockIdx.y * BM, bn0 = blockIdx.x * BN;

  // staging coordinates (per thread)
  const int xrow = t >> 3;            // +p*32, p=0..3
  const int xc4  = (t & 7) << 2;      // float column
  const int wrow = t >> 2;            // +p*64, p=0..1
  const int wc8  = (t & 3) << 3;      // bf16 column

  f32x8 acc[4][2];
  for (int i = 0; i < 4; ++i)
    for (int j = 0; j < 2; ++j)
      for (int r = 0; r < 8; ++r) acc[i][j][r] = 0.f;

  float4 xr[4];

  // ---- stage tile 0 ----
  #pragma unroll
  for (int p = 0; p < 4; ++p)
    xr[p] = *(const float4*)(&X[(size_t)(bm0 + xrow + p * 32) * DIM_K + xc4]);
  #pragma unroll
  for (int p = 0; p < 2; ++p)
    async_load_b128(lds_addr(&Ws[0][(wrow + p * 64) * LDP + wc8]),
                    &Wb[(size_t)(bn0 + wrow + p * 64) * DIM_K + wc8]);
  #pragma unroll
  for (int p = 0; p < 4; ++p) {
    bf16x4 bv;
    bv[0] = (__bf16)xr[p].x; bv[1] = (__bf16)xr[p].y;
    bv[2] = (__bf16)xr[p].z; bv[3] = (__bf16)xr[p].w;
    *(bf16x4*)(&Xs[0][(xrow + p * 32) * LDP + xc4]) = bv;
  }
  wait_async0();
  __syncthreads();

  #pragma unroll 1
  for (int kt = 0; kt < NT; ++kt) {
    const int cur = kt & 1, nxt = cur ^ 1;
    const int k1 = (kt + 1) * BK;
    const bool more = (kt + 1 < NT);

    // issue next tile's loads before computing the current one
    if (more) {
      #pragma unroll
      for (int p = 0; p < 4; ++p)
        xr[p] = *(const float4*)(&X[(size_t)(bm0 + xrow + p * 32) * DIM_K + k1 + xc4]);
      #pragma unroll
      for (int p = 0; p < 2; ++p)
        async_load_b128(lds_addr(&Ws[nxt][(wrow + p * 64) * LDP + wc8]),
                        &Wb[(size_t)(bn0 + wrow + p * 64) * DIM_K + k1 + wc8]);
    }

    // compute current tile
    bf16x16 afrag[4], bfrag[2];
    #pragma unroll
    for (int i = 0; i < 4; ++i)
      afrag[i] = ld_frag(&Xs[cur][(wm * 64 + i * 16 + l16) * LDP + lhalf * 8], 16);
    #pragma unroll
    for (int j = 0; j < 2; ++j)
      bfrag[j] = ld_frag(&Ws[cur][(wn * 32 + j * 16 + l16) * LDP + lhalf * 8], 16);

    #pragma unroll
    for (int i = 0; i < 4; ++i)
      #pragma unroll
      for (int j = 0; j < 2; ++j)
        acc[i][j] = __builtin_amdgcn_wmma_f32_16x16x32_bf16(
            false, afrag[i], false, bfrag[j], (short)0, acc[i][j], false, false);

    // finish next x tile: convert + park in LDS
    if (more) {
      #pragma unroll
      for (int p = 0; p < 4; ++p) {
        bf16x4 bv;
        bv[0] = (__bf16)xr[p].x; bv[1] = (__bf16)xr[p].y;
        bv[2] = (__bf16)xr[p].z; bv[3] = (__bf16)xr[p].w;
        *(bf16x4*)(&Xs[nxt][(xrow + p * 32) * LDP + xc4]) = bv;
      }
    }

    wait_async0();       // ASYNCcnt not covered by __syncthreads()
    __syncthreads();
  }

  // ---- epilogue: + bias, streaming (non-temporal) stores ----
  #pragma unroll
  for (int i = 0; i < 4; ++i) {
    int m = bm0 + wm * 64 + i * 16 + lhalf * 8;
    #pragma unroll
    for (int j = 0; j < 2; ++j) {
      int n = bn0 + wn * 32 + j * 16 + l16;
      float bv = bias[n];
      #pragma unroll
      for (int r = 0; r < 8; ++r)
        __builtin_nontemporal_store(acc[i][j][r] + bv,
                                    &Out[(size_t)(m + r) * DIM_N + n]);
    }
  }
}

extern "C" void kernel_launch(void* const* d_in, const int* in_sizes, int n_in,
                              void* d_out, int out_size, void* d_ws, size_t ws_size,
                              hipStream_t stream) {
  const float* x      = (const float*)d_in[0];   // [4,2048,4096]
  const float* W      = (const float*)d_in[1];   // [4096,4096]
  const float* bias   = (const float*)d_in[2];   // [4096]
  const float* A      = (const float*)d_in[3];   // [16,16,4096]
  const float* Bfac   = (const float*)d_in[4];   // [16,4096,16]
  const float* coeffs = (const float*)d_in[5];   // [16]
  const float* scales = (const float*)d_in[6];   // [16]
  float* out = (float*)d_out;

  // workspace layout (bf16): W_eff [4096*4096] | Bcat [4096*256] | At [4096*256]
  u16* Wb = (u16*)d_ws;
  u16* Bc = Wb + (size_t)DIM_N * DIM_K;
  u16* At = Bc + (size_t)DIM_N * RANKK;

  prep_factors<<<1024, 256, 0, stream>>>(A, Bfac, coeffs, scales, Bc, At);
  fold_wmma<<<dim3(DIM_K / BN, DIM_N / BM), 256, 0, stream>>>(Bc, At, W, Wb);
  main_wmma<<<dim3(DIM_N / BN, DIM_M / BM), 256, 0, stream>>>(x, Wb, bias, out);
}
